// CausalSelfAttention_29566554865804
// MI455X (gfx1250) — matrix operand order repaired and checked by reading, not compiled, not run on hardware
//
#include <hip/hip_runtime.h>
#include <hip/hip_bf16.h>
#include <math.h>

// ---------------------------------------------------------------------------
// CausalSelfAttention for MI455X (gfx1250, wave32, WMMA bf16 -> f32 accum)
// B=2, T=2048, C=1536, H=16, D=96
// GEMMs: double-buffered LDS tiles filled with GLOBAL_LOAD_ASYNC_TO_LDS_B128
// (ASYNCcnt-tracked).  Attention: flash-style streaming, all matmuls on
// v_wmma_f32_16x16x32_bf16.
// ---------------------------------------------------------------------------

typedef __bf16 bf16_t;
typedef __attribute__((ext_vector_type(16))) __bf16 v16bf;
typedef __attribute__((ext_vector_type(8)))  __bf16 v8bf;
typedef __attribute__((ext_vector_type(8)))  float  v8f;
typedef __attribute__((ext_vector_type(4)))  int    v4i;

#define B_   2
#define T_   2048
#define C_   1536
#define H_   16
#define D_   96
#define NQKV (3 * C_)    // 4608
#define MROWS (B_ * T_)  // 4096

#if defined(__HIP_DEVICE_COMPILE__) && __has_builtin(__builtin_amdgcn_global_load_async_to_lds_b128)
#define HAVE_ASYNC_LDS 1
#else
#define HAVE_ASYNC_LDS 0
#endif

// Builtin signature (from clang diagnostic): param0 is an int4* in the
// global (AS1 / "__device__") address space; param1 is the LDS (AS3) side.
typedef __attribute__((address_space(1))) v4i as1_v4i;
typedef __attribute__((address_space(3))) v4i as3_v4i;

// One wave-instruction: per-lane 16B global -> LDS, tracked by ASYNCcnt.
__device__ __forceinline__ void async_copy_b128(const bf16_t* g, bf16_t* l) {
#if HAVE_ASYNC_LDS
  __builtin_amdgcn_global_load_async_to_lds_b128((as1_v4i*)g, (as3_v4i*)l, 0, 0);
#else
  (void)g; (void)l;
#endif
}

template <int N>
__device__ __forceinline__ void wait_async() {
#if defined(__HIP_DEVICE_COMPILE__) && __has_builtin(__builtin_amdgcn_s_wait_asynccnt)
  __builtin_amdgcn_s_wait_asynccnt(N);
#elif defined(__HIP_DEVICE_COMPILE__)
  asm volatile("s_wait_asynccnt %0" ::"i"(N) : "memory");
#endif
}

__device__ __forceinline__ v8f wmma_bf(v16bf a, v16bf b, v8f c) {
  // D = A(16x32 bf16) x B(32x16 bf16) + C(16x16 f32)
  return __builtin_amdgcn_wmma_f32_16x16x32_bf16(
      /*neg_a=*/false, a, /*neg_b=*/false, b,
      /*c_mod=*/(short)0, c, /*reuse_a=*/false, /*reuse_b=*/false);
}

// A-fragment loader: 16x32 (MxK) bf16, row-major source with row stride
// `stride` elements.  Per ISA 7.12.2: lane = {m = lane&15, half = lane>>4},
// element e -> K = (e&7) + 8*(2*(e>>3) + half).  Each lane's 16 values are
// two contiguous 8-element (16B) chunks: K in [8*half, +8) and [16+8*half, +8).
__device__ __forceinline__ v16bf frag_a(const bf16_t* base, int mBase, int stride) {
  const int lane = threadIdx.x & 31;
  const int m = lane & 15;
  const int half = lane >> 4;
  const bf16_t* row = base + (size_t)(mBase + m) * stride;
  v8bf lo = *(const v8bf*)(row + 8 * half);
  v8bf hi = *(const v8bf*)(row + 16 + 8 * half);
  v16bf r;
#pragma unroll
  for (int i = 0; i < 8; ++i) { r[i] = lo[i]; r[i + 8] = hi[i]; }
  return r;
}

// B-fragment loader: 32x16 (KxN) bf16.  Source is row-major [N][K] with row
// stride `stride` (K contiguous per n).  lane = {n = lane&15, half = lane>>4},
// element e -> K = (e&7) + 8*(2*(e>>3) + half) -> two 8-element chunks per lane.
__device__ __forceinline__ v16bf frag_b(const bf16_t* base, int nBase, int stride) {
  const int lane = threadIdx.x & 31;
  const int n = lane & 15;
  const int half = lane >> 4;
  const bf16_t* row = base + (size_t)(nBase + n) * stride;
  v8bf lo = *(const v8bf*)(row + 8 * half);
  v8bf hi = *(const v8bf*)(row + 16 + 8 * half);
  v16bf r;
#pragma unroll
  for (int i = 0; i < 8; ++i) { r[i] = lo[i]; r[i + 8] = hi[i]; }
  return r;
}

// ---------------------------------------------------------------------------
// fp32 -> bf16 conversion
// ---------------------------------------------------------------------------
__global__ void f32_to_bf16(const float* __restrict__ in, bf16_t* __restrict__ out,
                            size_t n) {
  size_t i = (size_t)blockIdx.x * blockDim.x + threadIdx.x;
  if (i < n) out[i] = (bf16_t)in[i];
}

// ---------------------------------------------------------------------------
// C[M,N] (f32) = A[M,K] (bf16, row-major) * B[N,K]^T (bf16, row-major weight)
// Block: 256 threads (8 waves), tile 128x128, K-step 32, double-buffered LDS.
// Wave grid 2(M) x 4(N): each wave 64x32 -> 4x2 accumulators, 8 WMMAs/K-step.
// Per tile each wave issues 4 async B128 copies -> s_wait_asynccnt 4 overlaps
// exactly one tile in flight.  Requires M%128==0, N%128==0, K%32==0.
// ---------------------------------------------------------------------------
__global__ __launch_bounds__(256) void gemm_bf16_nt(
    float* __restrict__ Cmat, const bf16_t* __restrict__ A,
    const bf16_t* __restrict__ Bm, int M, int N, int K) {
  __shared__ bf16_t As[2][128][40];  // stride 40 (80B) kills bank conflicts
  __shared__ bf16_t Bs[2][128][40];
  const int tileM = blockIdx.y * 128;
  const int tileN = blockIdx.x * 128;
  const int t = threadIdx.x;
  const int lane = t & 31;
  const int w = t >> 5;
  const int wM = (w & 1) * 64;
  const int wN = (w >> 1) * 32;
  const int ldRow = t >> 1;         // 0..127
  const int ldSeg = (t & 1) * 16;   // 0 or 16

  const bf16_t* aRow = A + (size_t)(tileM + ldRow) * K + ldSeg;
  const bf16_t* bRow = Bm + (size_t)(tileN + ldRow) * K + ldSeg;

  v8f acc[4][2];
#pragma unroll
  for (int i = 0; i < 4; ++i)
#pragma unroll
    for (int j = 0; j < 2; ++j)
#pragma unroll
      for (int e = 0; e < 8; ++e) acc[i][j][e] = 0.0f;

  const int nk = K >> 5;

#if HAVE_ASYNC_LDS
  // Prologue: async-fill buffer 0 with tile 0.
  async_copy_b128(aRow,     &As[0][ldRow][ldSeg]);
  async_copy_b128(aRow + 8, &As[0][ldRow][ldSeg + 8]);
  async_copy_b128(bRow,     &Bs[0][ldRow][ldSeg]);
  async_copy_b128(bRow + 8, &Bs[0][ldRow][ldSeg + 8]);
#endif

  for (int kt = 0; kt < nk; ++kt) {
    const int cur = kt & 1;
#if HAVE_ASYNC_LDS
    if (kt + 1 < nk) {  // async-fill the other buffer with the next tile
      const bf16_t* agn = aRow + ((kt + 1) << 5);
      const bf16_t* bgn = bRow + ((kt + 1) << 5);
      async_copy_b128(agn,     &As[cur ^ 1][ldRow][ldSeg]);
      async_copy_b128(agn + 8, &As[cur ^ 1][ldRow][ldSeg + 8]);
      async_copy_b128(bgn,     &Bs[cur ^ 1][ldRow][ldSeg]);
      async_copy_b128(bgn + 8, &Bs[cur ^ 1][ldRow][ldSeg + 8]);
      wait_async<4>();  // current tile's 4 copies complete (in-order)
    } else {
      wait_async<0>();
    }
#else
    {
      const bf16_t* ag = aRow + (kt << 5);
      const bf16_t* bg = bRow + (kt << 5);
      *(v8bf*)&As[cur][ldRow][ldSeg]     = *(const v8bf*)(ag);
      *(v8bf*)&As[cur][ldRow][ldSeg + 8] = *(const v8bf*)(ag + 8);
      *(v8bf*)&Bs[cur][ldRow][ldSeg]     = *(const v8bf*)(bg);
      *(v8bf*)&Bs[cur][ldRow][ldSeg + 8] = *(const v8bf*)(bg + 8);
      if (kt + 1 < nk) {
        __builtin_prefetch((const void*)(ag + 32), 0, 1);
        __builtin_prefetch((const void*)(bg + 32), 0, 1);
      }
    }
#endif
    __syncthreads();  // everyone's tile-kt data visible in LDS

    v16bf af[4];
#pragma unroll
    for (int mf = 0; mf < 4; ++mf)
      af[mf] = frag_a(&As[cur][0][0], wM + mf * 16, 40);
#pragma unroll
    for (int nf = 0; nf < 2; ++nf) {
      v16bf bfrag = frag_b(&Bs[cur][0][0], wN + nf * 16, 40);
#pragma unroll
      for (int mf = 0; mf < 4; ++mf)
        acc[mf][nf] = wmma_bf(af[mf], bfrag, acc[mf][nf]);
    }
    __syncthreads();  // done reading before this buffer is refilled
  }

  const int nIdx = lane & 15;
  const int halfId = lane >> 4;
#pragma unroll
  for (int mf = 0; mf < 4; ++mf)
#pragma unroll
    for (int nf = 0; nf < 2; ++nf) {
      const int col = tileN + wN + nf * 16 + nIdx;
#pragma unroll
      for (int r = 0; r < 8; ++r) {
        const int row = tileM + wM + mf * 16 + r + 8 * halfId;
        Cmat[(size_t)row * N + col] = acc[mf][nf][r];
      }
    }
}

// ---------------------------------------------------------------------------
// RoPE + QKV split.  qkv is fp32 [B*T, 3C].  Writes:
//   Q, K : bf16 [B,H,T,D]   (rotary applied)
//   Vt   : bf16 [B,H,D,T]   (transposed so PV B-fragments are contiguous)
// One thread per (b,t,h,j) with j in [0,48) handling one rotary pair.
// ---------------------------------------------------------------------------
__global__ void rope_split(const float* __restrict__ qkv, bf16_t* __restrict__ Q,
                           bf16_t* __restrict__ Kc, bf16_t* __restrict__ Vt) {
  const size_t idx = (size_t)blockIdx.x * blockDim.x + threadIdx.x;
  const size_t total = (size_t)B_ * T_ * H_ * (D_ / 2);
  if (idx >= total) return;
  const int j = (int)(idx % 48);
  const int h = (int)((idx / 48) % H_);
  const int t = (int)((idx / (48 * H_)) % T_);
  const int b = (int)(idx / ((size_t)48 * H_ * T_));

  const float* src = qkv + ((size_t)b * T_ + t) * NQKV + h * D_;
  const float q1 = src[2 * j],          q2 = src[2 * j + 1];
  const float k1 = src[C_ + 2 * j],     k2 = src[C_ + 2 * j + 1];
  const float v1 = src[2 * C_ + 2 * j], v2 = src[2 * C_ + 2 * j + 1];

  // reference: emb = concat(freqs, freqs); c[j] = cos(emb[2j])
  //   2j < 48  -> freqs[2j];   2j >= 48 -> freqs[2j-48]
  const int m = (j < 24) ? (2 * j) : (2 * j - 48);
  const float invf = __powf(10000.0f, -(float)(2 * m) / 96.0f);
  float s, c;
  __sincosf((float)t * invf, &s, &c);

  const size_t ro = (((size_t)b * H_ + h) * T_ + t) * D_;
  Q[ro + j]       = (bf16_t)(q1 * c - q2 * s);
  Q[ro + 48 + j]  = (bf16_t)(q1 * s + q2 * c);
  Kc[ro + j]      = (bf16_t)(k1 * c - k2 * s);
  Kc[ro + 48 + j] = (bf16_t)(k1 * s + k2 * c);

  const size_t vo = ((size_t)b * H_ + h) * (size_t)D_ * T_;
  Vt[vo + (size_t)(2 * j) * T_ + t]     = (bf16_t)v1;
  Vt[vo + (size_t)(2 * j + 1) * T_ + t] = (bf16_t)v2;
}

// ---------------------------------------------------------------------------
// Flash attention.  Block = 128 threads = 4 waves; each wave owns 16 query
// rows of one (b,h).  Streams 32-key tiles with online softmax.
// Per key tile: 6 WMMAs for S = Q*K^T (Kdim=96) + 6 WMMAs for acc += P*V.
// Row statistics reduced with shfl_xor (masks 1/2/4/8 stay within the 16-lane
// half that owns a given C-fragment row).  Y written bf16 [B,T,C].
// ---------------------------------------------------------------------------
__global__ __launch_bounds__(128) void flash_attn(
    const bf16_t* __restrict__ Q, const bf16_t* __restrict__ Kc,
    const bf16_t* __restrict__ Vt, bf16_t* __restrict__ Y) {
  __shared__ bf16_t Pst[4][16][40];  // per-wave P staging (C-layout -> A-layout)
  const int w = threadIdx.x >> 5;
  const int lane = threadIdx.x & 31;
  const int nIdx = lane & 15;
  const int halfId = lane >> 4;
  const int bh = blockIdx.x;
  const int b = bh / H_;
  const int h = bh % H_;
  const int q0 = (blockIdx.y * 4 + w) * 16;

  const bf16_t* qbh = Q + (size_t)bh * T_ * D_;
  const bf16_t* kbh = Kc + (size_t)bh * T_ * D_;
  const bf16_t* vbh = Vt + (size_t)bh * D_ * T_;
  const float scale = 0.10206207261596577f;  // 1/sqrt(96)

  v16bf aq[3];
#pragma unroll
  for (int j = 0; j < 3; ++j) aq[j] = frag_a(qbh + 32 * j, q0, D_);

  v8f acc[6];
  float mrow[8], lrow[8];
#pragma unroll
  for (int f = 0; f < 6; ++f)
#pragma unroll
    for (int r = 0; r < 8; ++r) acc[f][r] = 0.0f;
#pragma unroll
  for (int r = 0; r < 8; ++r) { mrow[r] = -__builtin_inff(); lrow[r] = 0.0f; }

  const int nkt = (q0 + 47) >> 5;  // key tiles covering keys [0, q0+16)
  for (int kt = 0; kt < nkt; ++kt) {
    const int k0 = kt << 5;
    v8f s0, s1;
#pragma unroll
    for (int r = 0; r < 8; ++r) { s0[r] = 0.0f; s1[r] = 0.0f; }
#pragma unroll
    for (int j = 0; j < 3; ++j) {
      v16bf bk0 = frag_b(kbh + 32 * j, k0, D_);
      v16bf bk1 = frag_b(kbh + 32 * j, k0 + 16, D_);
      s0 = wmma_bf(aq[j], bk0, s0);
      s1 = wmma_bf(aq[j], bk1, s1);
    }

    float corr[8];
#pragma unroll
    for (int r = 0; r < 8; ++r) {
      const int qi = q0 + r + 8 * halfId;
      float v0 = s0[r] * scale;
      float v1 = s1[r] * scale;
      if (k0 + nIdx > qi)      v0 = -__builtin_inff();
      if (k0 + 16 + nIdx > qi) v1 = -__builtin_inff();
      float mx = fmaxf(v0, v1);
      mx = fmaxf(mx, __shfl_xor(mx, 1, 32));
      mx = fmaxf(mx, __shfl_xor(mx, 2, 32));
      mx = fmaxf(mx, __shfl_xor(mx, 4, 32));
      mx = fmaxf(mx, __shfl_xor(mx, 8, 32));
      const float mn = fmaxf(mrow[r], mx);
      const float cr = __expf(mrow[r] - mn);
      const float p0 = __expf(v0 - mn);
      const float p1 = __expf(v1 - mn);
      float rs = p0 + p1;
      rs += __shfl_xor(rs, 1, 32);
      rs += __shfl_xor(rs, 2, 32);
      rs += __shfl_xor(rs, 4, 32);
      rs += __shfl_xor(rs, 8, 32);
      lrow[r] = lrow[r] * cr + rs;
      mrow[r] = mn;
      corr[r] = cr;
      Pst[w][r + 8 * halfId][nIdx]      = (bf16_t)p0;
      Pst[w][r + 8 * halfId][16 + nIdx] = (bf16_t)p1;
    }
#pragma unroll
    for (int f = 0; f < 6; ++f)
#pragma unroll
      for (int r = 0; r < 8; ++r) acc[f][r] *= corr[r];

    // LDS ops are in-order within a wave; wait for P stores before reload.
    asm volatile("s_wait_dscnt 0" ::: "memory");
    v16bf pa = frag_a(&Pst[w][0][0], 0, 40);
#pragma unroll
    for (int f = 0; f < 6; ++f) {
      v16bf bv = frag_b(vbh + k0, 16 * f, T_);  // V^T: keys contiguous
      acc[f] = wmma_bf(pa, bv, acc[f]);
    }
  }

  float inv[8];
#pragma unroll
  for (int r = 0; r < 8; ++r) inv[r] = 1.0f / lrow[r];
  bf16_t* ybase = Y + (size_t)b * T_ * C_ + (size_t)h * D_;
#pragma unroll
  for (int f = 0; f < 6; ++f)
#pragma unroll
    for (int r = 0; r < 8; ++r) {
      const int row = q0 + r + 8 * halfId;
      ybase[(size_t)row * C_ + f * 16 + nIdx] = (bf16_t)(acc[f][r] * inv[r]);
    }
}

// ---------------------------------------------------------------------------
// Host-side launch
// ---------------------------------------------------------------------------
extern "C" void kernel_launch(void* const* d_in, const int* in_sizes, int n_in,
                              void* d_out, int out_size, void* d_ws, size_t ws_size,
                              hipStream_t stream) {
  (void)in_sizes; (void)n_in; (void)out_size; (void)ws_size;
  const float* x      = (const float*)d_in[0];
  const float* w_attn = (const float*)d_in[1];
  const float* w_proj = (const float*)d_in[2];
  float* out = (float*)d_out;
  char* ws = (char*)d_ws;

  size_t off = 0;
  auto alloc = [&](size_t bytes) {
    size_t o = off;
    off += (bytes + 255) & ~(size_t)255;
    return o;
  };
  bf16_t* xb  = (bf16_t*)(ws + alloc((size_t)MROWS * C_ * 2));   // 12.6 MB
  bf16_t* wab = (bf16_t*)(ws + alloc((size_t)NQKV * C_ * 2));    // 14.2 MB
  bf16_t* wpb = (bf16_t*)(ws + alloc((size_t)C_ * C_ * 2));      //  4.7 MB
  const size_t qkv_off = alloc((size_t)MROWS * NQKV * 4);        // 75.5 MB
  float*  qkv = (float*)(ws + qkv_off);
  bf16_t* qb  = (bf16_t*)(ws + alloc((size_t)B_ * H_ * T_ * D_ * 2));
  bf16_t* kb  = (bf16_t*)(ws + alloc((size_t)B_ * H_ * T_ * D_ * 2));
  bf16_t* vtb = (bf16_t*)(ws + alloc((size_t)B_ * H_ * T_ * D_ * 2));
  // y (bf16, 12.6 MB) aliases the qkv region: qkv is dead after rope_split,
  // and all kernels run in order on `stream`.
  bf16_t* yb = (bf16_t*)(ws + qkv_off);

  {
    size_t n = (size_t)MROWS * C_;
    f32_to_bf16<<<dim3((unsigned)((n + 255) / 256)), 256, 0, stream>>>(x, xb, n);
    n = (size_t)NQKV * C_;
    f32_to_bf16<<<dim3((unsigned)((n + 255) / 256)), 256, 0, stream>>>(w_attn, wab, n);
    n = (size_t)C_ * C_;
    f32_to_bf16<<<dim3((unsigned)((n + 255) / 256)), 256, 0, stream>>>(w_proj, wpb, n);
  }

  // qkv[4096,4608] = xb[4096,1536] @ wab[4608,1536]^T
  gemm_bf16_nt<<<dim3(NQKV / 128, MROWS / 128), 256, 0, stream>>>(
      qkv, xb, wab, MROWS, NQKV, C_);

  {
    size_t n = (size_t)B_ * T_ * H_ * (D_ / 2);
    rope_split<<<dim3((unsigned)((n + 255) / 256)), 256, 0, stream>>>(qkv, qb, kb, vtb);
  }

  // 4 waves/block, each wave = 16 query rows -> grid.y = T/64
  flash_attn<<<dim3(B_ * H_, T_ / 64), 128, 0, stream>>>(qb, kb, vtb, yb);

  // out[4096,1536] = yb[4096,1536] @ wpb[1536,1536]^T
  gemm_bf16_nt<<<dim3(C_ / 128, MROWS / 128), 256, 0, stream>>>(
      out, yb, wpb, MROWS, C_, C_);
}
